// RecGNN_62629213110866
// MI455X (gfx1250) — matrix operand order repaired and checked by reading, not compiled, not run on hardware
//
#include <hip/hip_runtime.h>
#include <math.h>

// ---- problem constants ----
#define NB 16
#define NV 2048
#define NC 16
#define ND 64
#define NHID 100
#define NG 256      // 4*D gates
#define NSTEPS 32

typedef __attribute__((ext_vector_type(16))) __bf16        v16bf;
typedef __attribute__((ext_vector_type(8)))  float         v8f;
typedef __attribute__((ext_vector_type(16))) unsigned short v16us;
typedef __attribute__((ext_vector_type(8)))  unsigned int  v8u;

static __device__ __forceinline__ unsigned short f2bf(float f) {
  unsigned u = __float_as_uint(f);
  return (unsigned short)((u + 0x7FFFu + ((u >> 16) & 1u)) >> 16);
}

static __device__ __forceinline__ v16bf ld_bf16(const unsigned short* p0,
                                                const unsigned short* p1) {
  uint4 a = *(const uint4*)p0;
  uint4 b = *(const uint4*)p1;
  v8u r;
  r[0]=a.x; r[1]=a.y; r[2]=a.z; r[3]=a.w;
  r[4]=b.x; r[5]=b.y; r[6]=b.z; r[7]=b.w;
  return __builtin_bit_cast(v16bf, r);
}

static __device__ __forceinline__ v16bf cvt_f32x16(const float* p0, const float* p1) {
  float4 a0 = *(const float4*)p0;
  float4 a1 = *(const float4*)(p0 + 4);
  float4 b0 = *(const float4*)p1;
  float4 b1 = *(const float4*)(p1 + 4);
  v16us u;
  u[0]=f2bf(a0.x);  u[1]=f2bf(a0.y);  u[2]=f2bf(a0.z);  u[3]=f2bf(a0.w);
  u[4]=f2bf(a1.x);  u[5]=f2bf(a1.y);  u[6]=f2bf(a1.z);  u[7]=f2bf(a1.w);
  u[8]=f2bf(b0.x);  u[9]=f2bf(b0.y);  u[10]=f2bf(b0.z); u[11]=f2bf(b0.w);
  u[12]=f2bf(b1.x); u[13]=f2bf(b1.y); u[14]=f2bf(b1.z); u[15]=f2bf(b1.w);
  return __builtin_bit_cast(v16bf, u);
}

// lower 8 K-values from memory, upper 8 zero (K=16 padded to K=32)
static __device__ __forceinline__ v16bf cvt_f32x8_pad(const float* p0) {
  float4 a0 = *(const float4*)p0;
  float4 a1 = *(const float4*)(p0 + 4);
  v16us u;
  u[0]=f2bf(a0.x); u[1]=f2bf(a0.y); u[2]=f2bf(a0.z); u[3]=f2bf(a0.w);
  u[4]=f2bf(a1.x); u[5]=f2bf(a1.y); u[6]=f2bf(a1.z); u[7]=f2bf(a1.w);
  u[8]=0; u[9]=0; u[10]=0; u[11]=0; u[12]=0; u[13]=0; u[14]=0; u[15]=0;
  return __builtin_bit_cast(v16bf, u);
}

static __device__ __forceinline__ v8f wmma_bf(v16bf a, v16bf b, v8f c) {
  return __builtin_amdgcn_wmma_f32_16x16x32_bf16(false, a, false, b,
                                                 (short)0, c, false, false);
}

static __device__ __forceinline__ float sigf(float x) {
  return 1.0f / (1.0f + __expf(-x));
}

// ----------------------------------------------------------------------------
// one-time kernels
// ----------------------------------------------------------------------------
__global__ void k_init(const float* __restrict__ vh_in, const float* __restrict__ ch_in,
                       float* __restrict__ vh0, float* __restrict__ vm0,
                       float* __restrict__ ch0, float* __restrict__ cm0,
                       unsigned short* __restrict__ Sc) {
  size_t i = (size_t)blockIdx.x * 256 + threadIdx.x;
  if (i < (size_t)NB * NV * ND) { vh0[i] = vh_in[i]; vm0[i] = 0.0f; }
  if (i < (size_t)NB * NC * ND) { ch0[i] = ch_in[i]; cm0[i] = 0.0f; }
  if (i < (size_t)NB * ND * 32) { Sc[i] = 0; }
}

__global__ void k_cvt(const float* __restrict__ in, unsigned short* __restrict__ out, int n) {
  int i = blockIdx.x * 256 + threadIdx.x;
  if (i < n) out[i] = f2bf(in[i]);
}

// McT[b][c][v] = bf16(Mvc[b][v][c])
__global__ void k_mvcT(const float* __restrict__ Mvc, unsigned short* __restrict__ out) {
  size_t o = (size_t)blockIdx.x * 256 + threadIdx.x;
  if (o >= (size_t)NB * NC * NV) return;
  int v = (int)(o & (NV - 1));
  int c = (int)((o >> 11) & 15);
  int b = (int)(o >> 15);
  out[o] = f2bf(Mvc[(((size_t)b * NV + v) << 4) + c]);
}

// ----------------------------------------------------------------------------
// per-step small kernels
// ----------------------------------------------------------------------------
// c_msg MLP, output staged transposed+padded bf16: Sc[b][d][k<32], k=c (c>=16 stays 0)
__global__ __launch_bounds__(128) void k_cmsg(const float* __restrict__ ch,
                                              const float* __restrict__ W1, const float* __restrict__ b1,
                                              const float* __restrict__ W2, const float* __restrict__ b2,
                                              unsigned short* __restrict__ Sc) {
  int row = blockIdx.x;            // 0..255
  int b = row >> 4, c = row & 15;
  __shared__ float hid[NHID];
  const float* x = ch + (size_t)row * ND;
  int t = threadIdx.x;
  if (t < NHID) {
    float s = b1[t];
    const float* w = W1 + (size_t)t * ND;
    for (int k = 0; k < ND; ++k) s += x[k] * w[k];
    hid[t] = fmaxf(s, 0.0f);
  }
  __syncthreads();
  if (t < ND) {
    float s = b2[t];
    const float* w = W2 + (size_t)t * NHID;
    for (int j = 0; j < NHID; ++j) s += hid[j] * w[j];
    Sc[(((size_t)b * ND + t) << 5) + c] = f2bf(fmaxf(s, 0.0f));
  }
}

// v_msg MLP on vh[B-1], output staged transposed bf16: SV[d][v]
__global__ __launch_bounds__(128) void k_vmsg(const float* __restrict__ vh,
                                              const float* __restrict__ W1, const float* __restrict__ b1,
                                              const float* __restrict__ W2, const float* __restrict__ b2,
                                              unsigned short* __restrict__ SV) {
  int row = blockIdx.x;            // 0..2047
  __shared__ float hid[NHID];
  const float* x = vh + (((size_t)(NB - 1) * NV + row) << 6);
  int t = threadIdx.x;
  if (t < NHID) {
    float s = b1[t];
    const float* w = W1 + (size_t)t * ND;
    for (int k = 0; k < ND; ++k) s += x[k] * w[k];
    hid[t] = fmaxf(s, 0.0f);
  }
  __syncthreads();
  if (t < ND) {
    float s = b2[t];
    const float* w = W2 + (size_t)t * NHID;
    for (int j = 0; j < NHID; ++j) s += hid[j] * w[j];
    SV[((size_t)t << 11) + row] = f2bf(fmaxf(s, 0.0f));
  }
}

// stage S[b][d][v] = bf16(vh[b][v][d]) via LDS transpose (coalesced both sides)
__global__ __launch_bounds__(256) void k_stage_vh(const float* __restrict__ vh,
                                                  unsigned short* __restrict__ S) {
  int b = blockIdx.y;
  int v0 = blockIdx.x * 64;
  __shared__ float tile[64][65];
  for (int i = threadIdx.x; i < 64 * 64; i += 256) {
    int r = i >> 6, d = i & 63;
    tile[r][d] = vh[(((size_t)b * NV + v0 + r) << 6) + d];
  }
  __syncthreads();
  for (int i = threadIdx.x; i < 64 * 64; i += 256) {
    int d = i >> 6, vv = i & 63;
    S[(((size_t)b * ND + d) << 11) + v0 + vv] = f2bf(tile[vv][d]);
  }
}

// ----------------------------------------------------------------------------
// v-node update: agg_v (WMMA, K=2048) + agg_c (WMMA, K=16 padded) + LSTM (WMMA)
// one wave per 16-row tile; grid = (V/16, B)
// ----------------------------------------------------------------------------
__global__ __launch_bounds__(32) void k_vnode(
    const unsigned short* __restrict__ MvvB,   // [B][V][V] bf16 row-major
    const float* __restrict__ Mvc,             // [B][V][C] f32
    const unsigned short* __restrict__ ScB,    // [B][D][32] bf16 (c_msg^T, padded)
    const unsigned short* __restrict__ Svh,    // [B][D][V] bf16 (vh^T)
    const float* __restrict__ vh_in, const float* __restrict__ vm_in,
    const unsigned short* __restrict__ WihB,   // [256][128] bf16 row-major
    const unsigned short* __restrict__ WhhB,   // [256][64]  bf16 row-major
    const float* __restrict__ bih, const float* __restrict__ bhh,
    float* __restrict__ vh_out, float* __restrict__ vm_out) {
  const int b    = blockIdx.y;
  const int v0   = blockIdx.x << 4;
  const int lane = threadIdx.x;
  const int half = lane >> 4;
  const int m    = lane & 15;

  __shared__ float xbuf[16][136];   // [rows][2D] f32, padded row stride (32B-aligned)

  v8f zero = {};
  v8f accv[4], accc[4];
#pragma unroll
  for (int t = 0; t < 4; ++t) { accv[t] = zero; accc[t] = zero; }

  // ---- agg_v = Mvv[b] @ vh[b]  (M=16 tile, N=64, K=2048) ----
  const unsigned short* arow = MvvB + ((size_t)(b * NV + v0 + m) << 11);
  const unsigned short* sB   = Svh + ((size_t)b << 17);  // 64*2048
  for (int k0 = 0; k0 < NV; k0 += 32) {
    v16bf A = ld_bf16(arow + k0 + 8 * half, arow + k0 + 16 + 8 * half);
#pragma unroll
    for (int t = 0; t < 4; ++t) {
      const unsigned short* bp = sB + (((size_t)(16 * t + m)) << 11) + k0 + 8 * half;
      accv[t] = wmma_bf(A, ld_bf16(bp, bp + 16), accv[t]);
    }
  }

  // ---- agg_c = Mvc[b] @ c_msg[b]  (K=16 zero-padded to 32) ----
  {
    const float* crow = Mvc + ((size_t)(b * NV + v0 + m) << 4) + 8 * half;
    v16bf A = cvt_f32x8_pad(crow);
#pragma unroll
    for (int t = 0; t < 4; ++t) {
      const unsigned short* bp = ScB + (((size_t)(b * ND + 16 * t + m)) << 5) + 8 * half;
      accc[t] = wmma_bf(A, ld_bf16(bp, bp + 16), accc[t]);
    }
  }

  // ---- stash x = [agg_c | agg_v] (C-layout -> LDS row-major) ----
#pragma unroll
  for (int t = 0; t < 4; ++t)
#pragma unroll
    for (int r = 0; r < 8; ++r) {
      xbuf[r + 8 * half][16 * t + m]      = accc[t][r];
      xbuf[r + 8 * half][64 + 16 * t + m] = accv[t][r];
    }
  __syncthreads();

  // ---- gates = x @ Wih^T + h @ Whh^T + biases  (16x256) ----
  v8f g[16];
#pragma unroll
  for (int gi = 0; gi < 16; ++gi) g[gi] = zero;

#pragma unroll
  for (int kk = 0; kk < 4; ++kk) {            // K = 128
    int k0 = kk << 5;
    v16bf A = cvt_f32x16(&xbuf[m][k0 + 8 * half], &xbuf[m][k0 + 16 + 8 * half]);
#pragma unroll
    for (int gi = 0; gi < 16; ++gi) {
      const unsigned short* bp = WihB + (((size_t)(16 * gi + m)) << 7) + k0 + 8 * half;
      g[gi] = wmma_bf(A, ld_bf16(bp, bp + 16), g[gi]);
    }
  }
  const float* hrow = vh_in + ((size_t)(b * NV + v0 + m) << 6);
#pragma unroll
  for (int kk = 0; kk < 2; ++kk) {            // K = 64
    int k0 = kk << 5;
    v16bf A = cvt_f32x16(hrow + k0 + 8 * half, hrow + k0 + 16 + 8 * half);
#pragma unroll
    for (int gi = 0; gi < 16; ++gi) {
      const unsigned short* bp = WhhB + (((size_t)(16 * gi + m)) << 6) + k0 + 8 * half;
      g[gi] = wmma_bf(A, ld_bf16(bp, bp + 16), g[gi]);
    }
  }
#pragma unroll
  for (int gi = 0; gi < 16; ++gi) {
    float bv = bih[16 * gi + m] + bhh[16 * gi + m];
#pragma unroll
    for (int r = 0; r < 8; ++r) g[gi][r] += bv;
  }

  // ---- LSTM elementwise; gate order i,f,g,o ----
#pragma unroll
  for (int t = 0; t < 4; ++t)
#pragma unroll
    for (int r = 0; r < 8; ++r) {
      int row = v0 + r + 8 * half;
      int col = (t << 4) + m;
      size_t idx = ((size_t)(b * NV + row) << 6) + col;
      float iv = g[t][r], fv = g[4 + t][r], gv = g[8 + t][r], ov = g[12 + t][r];
      float cold = vm_in[idx];
      float cn = sigf(fv) * cold + sigf(iv) * tanhf(gv);
      float hn = sigf(ov) * tanhf(cn);
      vm_out[idx] = cn;
      vh_out[idx] = hn;
    }
}

// ----------------------------------------------------------------------------
// c-node update: rnn_c_in = McT[b] @ v_msg (WMMA, K=2048) + LSTM (WMMA)
// one wave per batch; grid = B
// ----------------------------------------------------------------------------
__global__ __launch_bounds__(32) void k_cnode(
    const unsigned short* __restrict__ McTB,   // [B][C][V] bf16 row-major
    const unsigned short* __restrict__ SV,     // [D][V] bf16 (v_msg^T)
    const float* __restrict__ ch_in, const float* __restrict__ cm_in,
    const unsigned short* __restrict__ WihB,   // [256][64] bf16
    const unsigned short* __restrict__ WhhB,   // [256][64] bf16
    const float* __restrict__ bih, const float* __restrict__ bhh,
    float* __restrict__ ch_out, float* __restrict__ cm_out) {
  const int b    = blockIdx.x;
  const int lane = threadIdx.x;
  const int half = lane >> 4;
  const int m    = lane & 15;

  __shared__ float xbuf[16][72];

  v8f zero = {};
  v8f acc[4];
#pragma unroll
  for (int t = 0; t < 4; ++t) acc[t] = zero;

  const unsigned short* arow = McTB + ((size_t)(b * NC + m) << 11);
  for (int k0 = 0; k0 < NV; k0 += 32) {
    v16bf A = ld_bf16(arow + k0 + 8 * half, arow + k0 + 16 + 8 * half);
#pragma unroll
    for (int t = 0; t < 4; ++t) {
      const unsigned short* bp = SV + (((size_t)(16 * t + m)) << 11) + k0 + 8 * half;
      acc[t] = wmma_bf(A, ld_bf16(bp, bp + 16), acc[t]);
    }
  }
#pragma unroll
  for (int t = 0; t < 4; ++t)
#pragma unroll
    for (int r = 0; r < 8; ++r)
      xbuf[r + 8 * half][16 * t + m] = acc[t][r];
  __syncthreads();

  v8f g[16];
#pragma unroll
  for (int gi = 0; gi < 16; ++gi) g[gi] = zero;

#pragma unroll
  for (int kk = 0; kk < 2; ++kk) {            // x @ Wih^T, K = 64
    int k0 = kk << 5;
    v16bf A = cvt_f32x16(&xbuf[m][k0 + 8 * half], &xbuf[m][k0 + 16 + 8 * half]);
#pragma unroll
    for (int gi = 0; gi < 16; ++gi) {
      const unsigned short* bp = WihB + (((size_t)(16 * gi + m)) << 6) + k0 + 8 * half;
      g[gi] = wmma_bf(A, ld_bf16(bp, bp + 16), g[gi]);
    }
  }
  const float* hrow = ch_in + ((size_t)(b * NC + m) << 6);
#pragma unroll
  for (int kk = 0; kk < 2; ++kk) {            // h @ Whh^T, K = 64
    int k0 = kk << 5;
    v16bf A = cvt_f32x16(hrow + k0 + 8 * half, hrow + k0 + 16 + 8 * half);
#pragma unroll
    for (int gi = 0; gi < 16; ++gi) {
      const unsigned short* bp = WhhB + (((size_t)(16 * gi + m)) << 6) + k0 + 8 * half;
      g[gi] = wmma_bf(A, ld_bf16(bp, bp + 16), g[gi]);
    }
  }
#pragma unroll
  for (int gi = 0; gi < 16; ++gi) {
    float bv = bih[16 * gi + m] + bhh[16 * gi + m];
#pragma unroll
    for (int r = 0; r < 8; ++r) g[gi][r] += bv;
  }
#pragma unroll
  for (int t = 0; t < 4; ++t)
#pragma unroll
    for (int r = 0; r < 8; ++r) {
      int row = r + 8 * half;
      int col = (t << 4) + m;
      size_t idx = ((size_t)(b * NC + row) << 6) + col;
      float iv = g[t][r], fv = g[4 + t][r], gv = g[8 + t][r], ov = g[12 + t][r];
      float cold = cm_in[idx];
      float cn = sigf(fv) * cold + sigf(iv) * tanhf(gv);
      float hn = sigf(ov) * tanhf(cn);
      cm_out[idx] = cn;
      ch_out[idx] = hn;
    }
}

// ----------------------------------------------------------------------------
extern "C" void kernel_launch(void* const* d_in, const int* in_sizes, int n_in,
                              void* d_out, int out_size, void* d_ws, size_t ws_size,
                              hipStream_t stream) {
  (void)in_sizes; (void)n_in; (void)out_size; (void)ws_size;

  const float* Mvv   = (const float*)d_in[0];
  const float* Mvc   = (const float*)d_in[1];
  const float* vh_in = (const float*)d_in[2];
  const float* ch_in = (const float*)d_in[3];
  const float* cW1 = (const float*)d_in[4];  const float* cb1 = (const float*)d_in[5];
  const float* cW2 = (const float*)d_in[6];  const float* cb2 = (const float*)d_in[7];
  const float* vW1 = (const float*)d_in[8];  const float* vb1 = (const float*)d_in[9];
  const float* vW2 = (const float*)d_in[10]; const float* vb2 = (const float*)d_in[11];
  const float* rvWih = (const float*)d_in[12]; const float* rvbih = (const float*)d_in[13];
  const float* rvWhh = (const float*)d_in[14]; const float* rvbhh = (const float*)d_in[15];
  const float* rcWih = (const float*)d_in[16]; const float* rcbih = (const float*)d_in[17];
  const float* rcWhh = (const float*)d_in[18]; const float* rcbhh = (const float*)d_in[19];

  // ---- workspace carve-out ----
  char* w = (char*)d_ws;
  auto take = [&](size_t bytes) -> char* {
    char* p = w;
    w += (bytes + 255) & ~(size_t)255;
    return p;
  };
  const size_t NVH = (size_t)NB * NV * ND;   // 2,097,152
  const size_t NCH = (size_t)NB * NC * ND;   // 16,384
  float* vhb[2] = {(float*)take(NVH * 4), (float*)take(NVH * 4)};
  float* vmb[2] = {(float*)take(NVH * 4), (float*)take(NVH * 4)};
  float* chb[2] = {(float*)take(NCH * 4), (float*)take(NCH * 4)};
  float* cmb[2] = {(float*)take(NCH * 4), (float*)take(NCH * 4)};
  unsigned short* MvvB  = (unsigned short*)take((size_t)NB * NV * NV * 2);  // 128 MB (< L2)
  unsigned short* McTB  = (unsigned short*)take((size_t)NB * NC * NV * 2);
  unsigned short* Svh   = (unsigned short*)take((size_t)NB * ND * NV * 2);
  unsigned short* SV    = (unsigned short*)take((size_t)ND * NV * 2);
  unsigned short* Sc    = (unsigned short*)take((size_t)NB * ND * 32 * 2);
  unsigned short* WihVB = (unsigned short*)take((size_t)NG * 2 * ND * 2);
  unsigned short* WhhVB = (unsigned short*)take((size_t)NG * ND * 2);
  unsigned short* WihCB = (unsigned short*)take((size_t)NG * ND * 2);
  unsigned short* WhhCB = (unsigned short*)take((size_t)NG * ND * 2);

  // ---- one-time prep (every call: deterministic, capture-safe) ----
  k_init<<<(unsigned)((NVH + 255) / 256), 256, 0, stream>>>(vh_in, ch_in, vhb[0], vmb[0],
                                                            chb[0], cmb[0], Sc);
  {
    int n = NB * NV * NV;  // 67,108,864
    k_cvt<<<(unsigned)((n + 255) / 256), 256, 0, stream>>>(Mvv, MvvB, n);
  }
  k_cvt<<<(NG * 2 * ND + 255) / 256, 256, 0, stream>>>(rvWih, WihVB, NG * 2 * ND);
  k_cvt<<<(NG * ND + 255) / 256, 256, 0, stream>>>(rvWhh, WhhVB, NG * ND);
  k_cvt<<<(NG * ND + 255) / 256, 256, 0, stream>>>(rcWih, WihCB, NG * ND);
  k_cvt<<<(NG * ND + 255) / 256, 256, 0, stream>>>(rcWhh, WhhCB, NG * ND);
  k_mvcT<<<(unsigned)(((size_t)NB * NC * NV + 255) / 256), 256, 0, stream>>>(Mvc, McTB);

  // ---- 32 recurrent steps, double-buffered state ----
  for (int t = 0; t < NSTEPS; ++t) {
    int s = t & 1, d = s ^ 1;
    k_cmsg<<<NB * NC, 128, 0, stream>>>(chb[s], cW1, cb1, cW2, cb2, Sc);
    k_vmsg<<<NV, 128, 0, stream>>>(vhb[s], vW1, vb1, vW2, vb2, SV);
    k_stage_vh<<<dim3(NV / 64, NB), 256, 0, stream>>>(vhb[s], Svh);
    float* vout = (t == NSTEPS - 1) ? (float*)d_out : vhb[d];
    k_vnode<<<dim3(NV / 16, NB), 32, 0, stream>>>(MvvB, Mvc, Sc, Svh, vhb[s], vmb[s],
                                                  WihVB, WhhVB, rvbih, rvbhh,
                                                  vout, vmb[d]);
    k_cnode<<<NB, 32, 0, stream>>>(McTB, SV, chb[s], cmb[s],
                                   WihCB, WhhCB, rcbih, rcbhh, chb[d], cmb[d]);
  }
}
